// SeparableAttention_35424890257735
// MI455X (gfx1250) — matrix operand order repaired and checked by reading, not compiled, not run on hardware
//
#include <hip/hip_runtime.h>
#include <hip/hip_bf16.h>

// ---------------------------------------------------------------------------
// CDNA5 (gfx1250) fused multi-head causal self-attention.
// f16 matrix inputs, f32 WMMA accumulation (v_wmma_f32_16x16x32_f16), f32
// softmax. Memory-bound -> f16 intermediates + async global->LDS staging.
// ---------------------------------------------------------------------------

typedef __attribute__((ext_vector_type(16))) _Float16 v16h;
typedef __attribute__((ext_vector_type(8)))  _Float16 v8h;
typedef __attribute__((ext_vector_type(8)))  float    v8f;
typedef __attribute__((__vector_size__(16))) int      v4i_;

#define OUT_LEN 256
#define SRC     64
#define BSZ     4
#define EMB     512
#define NHEAD   8
#define HDIM    64
#define SB      (SRC * BSZ)        // 256 flattened batch
#define BH      (SB * NHEAD)       // 2048 attention instances
#define NROWS   (OUT_LEN * SB)     // 65536 GEMM rows
#define FDIM    (3 * EMB)          // 1536 qkv columns
#define NEG_INF (-1e9f)

#ifndef __has_builtin
#define __has_builtin(x) 0
#endif
#if defined(__HIP_DEVICE_COMPILE__) && \
    __has_builtin(__builtin_amdgcn_global_load_async_to_lds_b128)
#define HAVE_ASYNC 1
#else
#define HAVE_ASYNC 0
#endif

// 16B global->LDS copy; async (ASYNCcnt-tracked) when available.
// Builtin signature (from compiler diagnostic): param0 = AS1 v4i*, then
// AS3 v4i*, i32 offset, i32 cpol.
static __device__ inline void async_cp16(_Float16* l, const _Float16* g) {
#if HAVE_ASYNC
  __builtin_amdgcn_global_load_async_to_lds_b128(
      (__attribute__((address_space(1))) v4i_*)g,
      (__attribute__((address_space(3))) v4i_*)l, /*offset=*/0, /*cpol=*/0);
#else
  *(uint4*)l = *(const uint4*)g;
#endif
}
static __device__ inline void wait_async() {
#if HAVE_ASYNC
#if __has_builtin(__builtin_amdgcn_s_wait_asynccnt)
  __builtin_amdgcn_s_wait_asynccnt(0);
#else
  asm volatile("s_wait_asynccnt 0x0" ::: "memory");
#endif
#endif
}

// WMMA D = A(16x32 f16) * B(32x16 f16) + C(16x16 f32)
static __device__ inline v8f wmma16(v16h a, v16h b, v8f c) {
  return __builtin_amdgcn_wmma_f32_16x16x32_f16(
      false, a, false, b, (short)0, c, false, false);
}

// A-matrix (16x32, 16-bit) k order used here (not AMD's table order): lane ln = row M,
// lanes 0-15 K=0..15, lanes 16-31 K=16..31 (matches B) -> one contiguous run.
static __device__ inline int kmapA(int e, int g) {
  return e + (g << 4);
}
// B-matrix (32x16) k order used here: lanes 0-15 K=0..15, lanes 16-31 K=16..31 -> one
// contiguous 32B run per lane.

// convert 16 f32 -> 16 f16 and store to LDS as two b128
static __device__ inline void cvt_store16(_Float16* dst, const float* src) {
  const float4* p = (const float4*)src;
#pragma unroll
  for (int h = 0; h < 2; ++h) {
    float4 f0 = p[2 * h], f1 = p[2 * h + 1];
    v8h v;
    v[0] = (_Float16)f0.x; v[1] = (_Float16)f0.y;
    v[2] = (_Float16)f0.z; v[3] = (_Float16)f0.w;
    v[4] = (_Float16)f1.x; v[5] = (_Float16)f1.y;
    v[6] = (_Float16)f1.z; v[7] = (_Float16)f1.w;
    *(v8h*)(dst + 8 * h) = v;
  }
}

// ---------------------------------------------------------------------------
// Kernel 1: qkv = X @ Win^T + b ; split into head-major f16 q (pre-scaled), k, v
// ---------------------------------------------------------------------------
__global__ __launch_bounds__(128) void qkv_proj_kernel(
    const float* __restrict__ X, const float* __restrict__ W,
    const float* __restrict__ bias,
    _Float16* __restrict__ qbuf, _Float16* __restrict__ kbuf,
    _Float16* __restrict__ vbuf) {
  __shared__ _Float16 Al[64 * 32];
  __shared__ _Float16 Bl[64 * 32];
  const int tid = threadIdx.x;
  const int w = tid >> 5, lane = tid & 31, g = lane >> 4, ln = lane & 15;
  const int m0 = blockIdx.x * 64;
  const int f0 = blockIdx.y * 64;
  const int sr = tid >> 1, sc = (tid & 1) * 16;   // staging row/col (16 f32)

  v8f acc[4];
#pragma unroll
  for (int t = 0; t < 4; ++t)
#pragma unroll
    for (int i = 0; i < 8; ++i) acc[t][i] = 0.0f;

  for (int kc = 0; kc < EMB / 32; ++kc) {
    __syncthreads();
    cvt_store16(&Al[sr * 32 + sc], &X[(size_t)(m0 + sr) * EMB + kc * 32 + sc]);
    cvt_store16(&Bl[sr * 32 + sc], &W[(size_t)(f0 + sr) * EMB + kc * 32 + sc]);
    if (kc + 1 < EMB / 32) {   // prefetch next K-chunk -> global_prefetch_b8
      __builtin_prefetch(&X[(size_t)(m0 + sr) * EMB + (kc + 1) * 32 + sc], 0, 0);
      __builtin_prefetch(&W[(size_t)(f0 + sr) * EMB + (kc + 1) * 32 + sc], 0, 0);
    }
    __syncthreads();
    v16h a;
#pragma unroll
    for (int e = 0; e < 16; ++e) a[e] = Al[(w * 16 + ln) * 32 + kmapA(e, g)];
#pragma unroll
    for (int t = 0; t < 4; ++t) {
      v16h b = *(const v16h*)&Bl[(t * 16 + ln) * 32 + g * 16];
      acc[t] = wmma16(a, b, acc[t]);
    }
  }

  // scatter-store into head-major layouts
#pragma unroll
  for (int t = 0; t < 4; ++t) {
    int f = f0 + t * 16 + ln;
    int which = f >> 9;              // 0=q 1=k 2=v
    int h = (f >> 6) & 7;
    int d = f & 63;
    float bv = bias[f];
#pragma unroll
    for (int i = 0; i < 8; ++i) {
      int r = m0 + w * 16 + i + 8 * g;
      int o = r >> 8, sb = r & 255;
      int bh = sb * NHEAD + h;
      float val = acc[t][i] + bv;
      if (which == 0) val *= 0.125f;  // HDIM^-0.5
      size_t off = ((size_t)bh * OUT_LEN + o) * HDIM + d;
      _Float16* dst = (which == 0) ? qbuf : (which == 1) ? kbuf : vbuf;
      dst[off] = (_Float16)val;
    }
  }
}

// ---------------------------------------------------------------------------
// Kernel 2: per (sb, 64-row q tile) block, loop all 8 heads:
//   async-stage Q/K/V tiles to LDS, S = q@k^T (+causal), f32 softmax
//   (shfl row-reduce), head-avg -> LDS, probs(f16) -> LDS, O = P@V via WMMA.
// Dynamic LDS (200 KB of the 320 KB WGP pool):
//   avg f32 64x256 | probs f16 64x256 | VT f16 64x256 | Kt 256x64 | Vt 256x64
//   | Qt 64x64
// ---------------------------------------------------------------------------
__global__ __launch_bounds__(128) void attention_kernel(
    const _Float16* __restrict__ qbuf, const _Float16* __restrict__ kbuf,
    const _Float16* __restrict__ vbuf, _Float16* __restrict__ attn_h,
    float* __restrict__ wavg_out) {
  extern __shared__ char smem[];
  float*    avg   = (float*)smem;                        // 65536 B
  _Float16* probs = (_Float16*)(smem + 65536);           // 32768 B
  _Float16* VT    = (_Float16*)(smem + 98304);           // 32768 B [d][key]
  _Float16* Kt    = (_Float16*)(smem + 131072);          // 32768 B [key][d]
  _Float16* Vt    = (_Float16*)(smem + 163840);          // 32768 B [key][d]
  _Float16* Qt    = (_Float16*)(smem + 196608);          // 8192  B [row][d]

  const int tid = threadIdx.x;
  const int w = tid >> 5, lane = tid & 31, g = lane >> 4, ln = lane & 15;
  const int obase = blockIdx.x * 64;
  const int sb = blockIdx.y;

  for (int j = tid; j < 64 * 256; j += 128) avg[j] = 0.0f;

  for (int h = 0; h < NHEAD; ++h) {
    const int bh = sb * NHEAD + h;
    const _Float16* kg = kbuf + (size_t)bh * OUT_LEN * HDIM;
    const _Float16* vg = vbuf + (size_t)bh * OUT_LEN * HDIM;
    const _Float16* qg = qbuf + (size_t)bh * OUT_LEN * HDIM + (size_t)obase * HDIM;

    __syncthreads();  // prior head's LDS reads done before re-staging
    // ---- async bulk copies: K tile 32KB, V tile 32KB, Q tile 8KB
#pragma unroll
    for (int j = 0; j < 16; ++j) {
      int c = j * 128 + tid;
      async_cp16(&Kt[c * 8], kg + c * 8);
      async_cp16(&Vt[c * 8], vg + c * 8);
    }
#pragma unroll
    for (int j = 0; j < 4; ++j) {
      int c = j * 128 + tid;
      async_cp16(&Qt[c * 8], qg + c * 8);
    }
    wait_async();
    __syncthreads();

    // ---- transpose V in LDS: VT[d][key] (vector loads, scalar stores)
    for (int it = 0; it < 16; ++it) {
      int idx8 = it * 128 + tid;
      int key = idx8 >> 3, d0 = (idx8 & 7) * 8;
      v8h vv = *(const v8h*)&Vt[key * 64 + d0];
#pragma unroll
      for (int q = 0; q < 8; ++q) VT[(d0 + q) * 256 + key] = vv[q];
    }

    // ---- scores: 16 rows/wave x 256 keys, K = HDIM = 64 (2 chunks of 32)
    v8f s[16];
#pragma unroll
    for (int t = 0; t < 16; ++t)
#pragma unroll
      for (int i = 0; i < 8; ++i) s[t][i] = 0.0f;

#pragma unroll
    for (int dc = 0; dc < 2; ++dc) {
      v16h a;
#pragma unroll
      for (int e = 0; e < 16; ++e)
        a[e] = Qt[(w * 16 + ln) * HDIM + dc * 32 + kmapA(e, g)];
#pragma unroll
      for (int t = 0; t < 16; ++t) {
        v16h b = *(const v16h*)&Kt[(t * 16 + ln) * HDIM + dc * 32 + g * 16];
        s[t] = wmma16(a, b, s[t]);
      }
    }

    // ---- causal mask + softmax (rows m = i + 8g live in a 16-lane half)
#pragma unroll
    for (int i = 0; i < 8; ++i) {
      const int qr = obase + w * 16 + i + 8 * g;
#pragma unroll
      for (int t = 0; t < 16; ++t) {
        int key = t * 16 + ln;
        if (key > qr) s[t][i] += NEG_INF;
      }
      float m_ = -3.4e38f;
#pragma unroll
      for (int t = 0; t < 16; ++t) m_ = fmaxf(m_, s[t][i]);
#pragma unroll
      for (int off = 8; off >= 1; off >>= 1)
        m_ = fmaxf(m_, __shfl_xor(m_, off, 16));
      float sum = 0.0f;
#pragma unroll
      for (int t = 0; t < 16; ++t) {
        float p = __expf(s[t][i] - m_);
        s[t][i] = p;
        sum += p;
      }
#pragma unroll
      for (int off = 8; off >= 1; off >>= 1) sum += __shfl_xor(sum, off, 16);
      float inv = 1.0f / sum;
      const int rowl = w * 16 + i + 8 * g;
#pragma unroll
      for (int t = 0; t < 16; ++t) {
        float p = s[t][i] * inv;
        int col = t * 16 + ln;
        probs[rowl * 256 + col] = (_Float16)p;
        avg[rowl * 256 + col] += p * 0.125f;  // mean over 8 heads
      }
    }
    __syncthreads();

    // ---- O = P @ V  (K = 256 keys, 8 chunks of 32; N = 4 d-tiles of 16)
    v8f o4[4];
#pragma unroll
    for (int t = 0; t < 4; ++t)
#pragma unroll
      for (int i = 0; i < 8; ++i) o4[t][i] = 0.0f;
#pragma unroll
    for (int ic = 0; ic < 8; ++ic) {
      v16h a;
#pragma unroll
      for (int e = 0; e < 16; ++e)
        a[e] = probs[(w * 16 + ln) * 256 + ic * 32 + kmapA(e, g)];
#pragma unroll
      for (int dt = 0; dt < 4; ++dt) {
        v16h b = *(const v16h*)&VT[(dt * 16 + ln) * 256 + ic * 32 + g * 16];
        o4[dt] = wmma16(a, b, o4[dt]);
      }
    }
    // store pre-out-proj activation rows r = o*SB + sb, col h*64 + d
#pragma unroll
    for (int dt = 0; dt < 4; ++dt)
#pragma unroll
      for (int i = 0; i < 8; ++i) {
        int qr = obase + w * 16 + i + 8 * g;
        attn_h[((size_t)qr * SB + sb) * EMB + h * HDIM + dt * 16 + ln] =
            (_Float16)o4[dt][i];
      }
  }

  __syncthreads();
  // averaged attention weights: [src,bsz,out,in] -> (sb*256 + o)*256 + i
  for (int j = tid; j < 64 * 256; j += 128) {
    int row = j >> 8, col = j & 255;
    wavg_out[((size_t)sb * OUT_LEN + obase + row) * OUT_LEN + col] = avg[j];
  }
}

// ---------------------------------------------------------------------------
// Kernel 3: out = attn_h(f16) @ Wout^T + b  ->  f32 [65536, 512]
// A tile staged with async global->LDS (already f16); W converted on the fly.
// ---------------------------------------------------------------------------
__global__ __launch_bounds__(128) void out_proj_kernel(
    const _Float16* __restrict__ A, const float* __restrict__ W,
    const float* __restrict__ bias, float* __restrict__ out) {
  __shared__ _Float16 Al[64 * 32];
  __shared__ _Float16 Bl[64 * 32];
  const int tid = threadIdx.x;
  const int w = tid >> 5, lane = tid & 31, g = lane >> 4, ln = lane & 15;
  const int m0 = blockIdx.x * 64;
  const int f0 = blockIdx.y * 64;
  const int sr = tid >> 1, sc = (tid & 1) * 16;

  v8f acc[4];
#pragma unroll
  for (int t = 0; t < 4; ++t)
#pragma unroll
    for (int i = 0; i < 8; ++i) acc[t][i] = 0.0f;

  for (int kc = 0; kc < EMB / 32; ++kc) {
    __syncthreads();
    // Al: 2048 halves = 256 x 16B chunks, 2 per thread (async)
#pragma unroll
    for (int j = 0; j < 2; ++j) {
      int c = j * 128 + tid;
      int r = c >> 2, c0 = (c & 3) * 8;
      async_cp16(&Al[r * 32 + c0],
                 &A[(size_t)(m0 + r) * EMB + kc * 32 + c0]);
    }
    cvt_store16(&Bl[sr * 32 + sc], &W[(size_t)(f0 + sr) * EMB + kc * 32 + sc]);
    if (kc + 1 < EMB / 32)
      __builtin_prefetch(&W[(size_t)(f0 + sr) * EMB + (kc + 1) * 32 + sc], 0, 0);
    wait_async();
    __syncthreads();
    v16h a;
#pragma unroll
    for (int e = 0; e < 16; ++e) a[e] = Al[(w * 16 + ln) * 32 + kmapA(e, g)];
#pragma unroll
    for (int t = 0; t < 4; ++t) {
      v16h b = *(const v16h*)&Bl[(t * 16 + ln) * 32 + g * 16];
      acc[t] = wmma16(a, b, acc[t]);
    }
  }
#pragma unroll
  for (int t = 0; t < 4; ++t) {
    int f = f0 + t * 16 + ln;
    float bv = bias[f];
#pragma unroll
    for (int i = 0; i < 8; ++i) {
      int r = m0 + w * 16 + i + 8 * g;
      out[(size_t)r * EMB + f] = acc[t][i] + bv;
    }
  }
}

// ---------------------------------------------------------------------------
extern "C" void kernel_launch(void* const* d_in, const int* in_sizes, int n_in,
                              void* d_out, int out_size, void* d_ws, size_t ws_size,
                              hipStream_t stream) {
  const float* query  = (const float*)d_in[0];
  const float* w_in   = (const float*)d_in[1];
  const float* b_in   = (const float*)d_in[2];
  const float* w_out  = (const float*)d_in[3];
  const float* b_out  = (const float*)d_in[4];
  // d_in[5] = attn_mask (causal -1e9 upper triangle) computed inline.

  // workspace partition (f16): q,k,v [2048][256][64]; attn_h [65536][512]
  const size_t HV = (size_t)BH * OUT_LEN * HDIM;  // 33,554,432 halves
  _Float16* qbuf   = (_Float16*)d_ws;
  _Float16* kbuf   = qbuf + HV;
  _Float16* vbuf   = kbuf + HV;
  _Float16* attn_h = vbuf + HV;

  float* out_attn = (float*)d_out;                          // [256,64,4,512]
  float* out_wavg = out_attn + (size_t)NROWS * EMB;         // [64,4,256,256]

  qkv_proj_kernel<<<dim3(NROWS / 64, FDIM / 64), 128, 0, stream>>>(
      query, w_in, b_in, qbuf, kbuf, vbuf);

  attention_kernel<<<dim3(OUT_LEN / 64, SB), 128, 204800, stream>>>(
      qbuf, kbuf, vbuf, attn_h, out_wavg);

  out_proj_kernel<<<dim3(NROWS / 64, EMB / 64), 128, 0, stream>>>(
      attn_h, w_out, b_out, out_attn);
}
